// Mamba_Encoder_67516885893472
// MI455X (gfx1250) — compile-verified, matrix operand order checked by reading, not run on hardware
//
#include <hip/hip_runtime.h>

// ---------------------------------------------------------------------------
// Types for CDNA5 WMMA (wave32, v_wmma_f32_16x16x32_bf16)
// ---------------------------------------------------------------------------
typedef __bf16 bf16;
typedef __attribute__((ext_vector_type(16))) __bf16 v16bf;
typedef __attribute__((ext_vector_type(8)))  __bf16 v8bf;
typedef __attribute__((ext_vector_type(8)))  float  v8f;

#define NT_TOKENS 32768   // BN * T = 64 * 512
#define SEQ_T     512
#define BN_SEQS   64

// ---------------------------------------------------------------------------
// f32 -> bf16 weight conversion
// ---------------------------------------------------------------------------
__global__ void f2bf_kernel(const float* __restrict__ s, bf16* __restrict__ d, int n) {
    int i = blockIdx.x * blockDim.x + threadIdx.x;
    if (i < n) d[i] = (bf16)s[i];
}

// ---------------------------------------------------------------------------
// Front-end: 4 dilated conv1d (M=32 -> 64 ch each, k=3, d=1,2,4,8), SAME pad,
// concatenated to 256 features, written token-major bf16 for the fuse GEMM.
// x layout: (BN, 32, T) f32.  out: (NT, 256) bf16.
// ---------------------------------------------------------------------------
__global__ __launch_bounds__(256)
void dilated_conv_kernel(const float* __restrict__ x,
                         const float* __restrict__ w0, const float* __restrict__ b0,
                         const float* __restrict__ w1, const float* __restrict__ b1,
                         const float* __restrict__ w2, const float* __restrict__ b2,
                         const float* __restrict__ w3, const float* __restrict__ b3,
                         bf16* __restrict__ out) {
    const int token = blockIdx.x;          // 0..NT-1
    const int c     = threadIdx.x;         // 0..255
    const int bn    = token >> 9;
    const int t     = token & (SEQ_T - 1);
    const int ci    = c >> 6;              // which conv branch
    const int oc    = c & 63;
    const float* w; const float* bb; int d;
    if      (ci == 0) { w = w0; bb = b0; d = 1; }
    else if (ci == 1) { w = w1; bb = b1; d = 2; }
    else if (ci == 2) { w = w2; bb = b2; d = 4; }
    else              { w = w3; bb = b3; d = 8; }
    const float* xb = x + (size_t)bn * 32 * SEQ_T;
    const float* wr = w + oc * (32 * 3);   // (64,32,3) OIH
    float s = bb[oc];
    #pragma unroll 4
    for (int cin = 0; cin < 32; ++cin) {
        const float* xr = xb + cin * SEQ_T;
        float acc = wr[cin * 3 + 1] * xr[t];
        if (t - d >= 0)    acc += wr[cin * 3 + 0] * xr[t - d];
        if (t + d < SEQ_T) acc += wr[cin * 3 + 2] * xr[t + d];
        s += acc;
    }
    out[(size_t)token * 256 + c] = (bf16)s;
}

// ---------------------------------------------------------------------------
// WMMA GEMM:  out[token, n] = act( sum_k A[token,k] * W[n,k] + bias[n] )
// A: (NT, K) bf16 row-major, W: (N, K) bf16 row-major (i.e. x @ W.T).
// Block = 256 threads = 8 waves; each wave computes a 16 x (JT*16) tile with
// JT accumulators sharing one A fragment per K-step.  Ping-pong software
// pipeline (2x unrolled): each buffer is redefined directly by its loads, so
// there is no register-rotation copy storm, and the s_wait_loadcnt waits stay
// partial (loads for the next stage overlap the WMMAs of the current stage).
// Requires K % 64 == 0.  act: 0 none, 1 relu, 2 silu.  outB/outF may be null.
// ---------------------------------------------------------------------------
template <int JT>
__global__ __launch_bounds__(256)
void gemm_bf16_wmma(const bf16* __restrict__ A, const bf16* __restrict__ W,
                    const float* __restrict__ bias,
                    bf16* __restrict__ outB, float* __restrict__ outF,
                    int N, int K, int act) {
    const int lane = threadIdx.x & 31;
    const int wave = threadIdx.x >> 5;
    const int l    = lane & 15;
    const int hi   = lane >> 4;                 // 0 or 1
    const int row0 = (blockIdx.y * 8 + wave) * 16;
    const int n0   = blockIdx.x * (JT * 16);

    v8f acc[JT];
    #pragma unroll
    for (int j = 0; j < JT; ++j)
        acc[j] = (v8f){0.f, 0.f, 0.f, 0.f, 0.f, 0.f, 0.f, 0.f};

    // 16-bit A 16x32 layout: lanes 0-15 hold K {0..7,16..23}; lanes 16-31 hold
    // K {8..15,24..31} (two 16B chunks per lane). B (K x 16) mirrored, lane = n.
    const bf16* aRow = A + (size_t)(row0 + l) * K + (hi ? 8 : 0);
    const bf16* wRow = W + (size_t)(n0  + l) * K + (hi ? 8 : 0);

    auto ldfrag = [](const bf16* p) -> v16bf {
        v8bf x0 = *(const v8bf*)(p);
        v8bf x1 = *(const v8bf*)(p + 16);
        return __builtin_shufflevector(x0, x1,
            0,1,2,3,4,5,6,7,8,9,10,11,12,13,14,15);
    };

    // ping buffer: fragments for k = 0
    v16bf a0 = ldfrag(aRow);
    v16bf b0[JT];
    #pragma unroll
    for (int j = 0; j < JT; ++j) b0[j] = ldfrag(wRow + (size_t)j * 16 * K);

    for (int k = 0; k < K; k += 64) {
        // pong buffer: fragments for k + 32 (always exists: K % 64 == 0)
        v16bf a1 = ldfrag(aRow + k + 32);
        v16bf b1[JT];
        #pragma unroll
        for (int j = 0; j < JT; ++j)
            b1[j] = ldfrag(wRow + (size_t)j * 16 * K + k + 32);

        // stage 1: consume ping (k)
        #pragma unroll
        for (int j = 0; j < JT; ++j)
            acc[j] = __builtin_amdgcn_wmma_f32_16x16x32_bf16(
                false, a0, false, b0[j], (short)0, acc[j], false, false);

        // refill ping with k + 64 (uniform guard; loads define the buffer)
        if (k + 64 < K) {
            a0 = ldfrag(aRow + k + 64);
            #pragma unroll
            for (int j = 0; j < JT; ++j)
                b0[j] = ldfrag(wRow + (size_t)j * 16 * K + k + 64);
        }

        // stage 2: consume pong (k + 32)
        #pragma unroll
        for (int j = 0; j < JT; ++j)
            acc[j] = __builtin_amdgcn_wmma_f32_16x16x32_bf16(
                false, a1, false, b1[j], (short)0, acc[j], false, false);
    }

    // C/D layout: lane n (0-15) cols; VGPR m -> row m (lanes 0-15) / m+8 (16-31)
    #pragma unroll
    for (int j = 0; j < JT; ++j) {
        const int col = n0 + j * 16 + l;
        const float bv = bias ? bias[col] : 0.f;
        #pragma unroll
        for (int m = 0; m < 8; ++m) {
            const int row = row0 + m + (hi ? 8 : 0);
            float v = acc[j][m] + bv;
            if (act == 1)      v = fmaxf(v, 0.f);
            else if (act == 2) v = v / (1.f + __expf(-v));
            const size_t o = (size_t)row * N + col;
            if (outF) outF[o] = v;
            if (outB) outB[o] = (bf16)v;
        }
    }
}

// ---------------------------------------------------------------------------
// Depthwise causal conv (k=4) + SiLU:  u[b,t,d] from xi = xz[:, 0:dI]
// xz: (NT, 2*dI) bf16.  u: (NT, dI) bf16.  DI is compile-time (512 or 1024).
// ---------------------------------------------------------------------------
template <int DI>
__global__ void dwconv_silu_kernel(const bf16* __restrict__ xz,
                                   const float* __restrict__ cw,  // (DI,1,4)
                                   const float* __restrict__ cb,
                                   bf16* __restrict__ u) {
    const int idx = blockIdx.x * blockDim.x + threadIdx.x;
    const int d     = idx & (DI - 1);
    const int token = idx / DI;           // DI power of two -> shift
    const int t     = token & (SEQ_T - 1);
    const float* w = cw + d * 4;
    float s = cb[d];
    #pragma unroll
    for (int k = 0; k < 4; ++k) {
        const int tt = t - 3 + k;
        if (tt >= 0)
            s += w[k] * (float)xz[(size_t)(token - 3 + k) * (2 * DI) + d];
    }
    s = s / (1.f + __expf(-s));                 // silu
    u[(size_t)token * DI + d] = (bf16)s;
}

// ---------------------------------------------------------------------------
// Fused selective scan. One block per sequence b, one thread per channel d.
// Fuses: dt = softplus(dtr . dt_proj_w[d] + b), recurrence over 16 states,
// y = (scan + u*D) * silu(z).  xdbl row (dtr | B | C) staged in LDS per step.
// ---------------------------------------------------------------------------
template <int DTR>
__global__ __launch_bounds__(1024, 1)
void scan_kernel(const float* __restrict__ xdbl,   // (NT, DTR+32)
                 const bf16*  __restrict__ xz,     // (NT, 2*dI), z at col dI+d
                 const bf16*  __restrict__ u,      // (NT, dI)
                 const float* __restrict__ Alog,   // (dI, 16)
                 const float* __restrict__ dtw,    // (dI, DTR)
                 const float* __restrict__ dtb,    // (dI)
                 const float* __restrict__ Dp,     // (dI)
                 bf16* __restrict__ y,             // (NT, dI)
                 int dI) {
    constexpr int XW = DTR + 32;
    const int b = blockIdx.x;
    const int d = threadIdx.x;
    __shared__ float sh[XW];

    float Ar[16];
    #pragma unroll
    for (int s = 0; s < 16; ++s) Ar[s] = -__expf(Alog[d * 16 + s]);
    float wdt[DTR];
    #pragma unroll
    for (int r = 0; r < DTR; ++r) wdt[r] = dtw[d * DTR + r];
    const float bdt = dtb[d];
    const float Dd  = Dp[d];
    float h[16];
    #pragma unroll
    for (int s = 0; s < 16; ++s) h[s] = 0.f;

    for (int t = 0; t < SEQ_T; ++t) {
        const size_t token = (size_t)b * SEQ_T + t;
        __syncthreads();
        if (threadIdx.x < XW) sh[threadIdx.x] = xdbl[token * XW + threadIdx.x];
        __syncthreads();

        float dt = bdt;
        #pragma unroll
        for (int r = 0; r < DTR; ++r) dt += wdt[r] * sh[r];
        dt = (dt > 20.f) ? dt : log1pf(__expf(dt));          // softplus

        const float uv = (float)u[token * dI + d];
        const float du = dt * uv;
        float yv = 0.f;
        #pragma unroll
        for (int s = 0; s < 16; ++s) {
            const float dA = __expf(dt * Ar[s]);
            h[s] = dA * h[s] + du * sh[DTR + s];             // B
            yv  += h[s] * sh[DTR + 16 + s];                  // C
        }
        const float zv = (float)xz[token * (size_t)(2 * dI) + dI + d];
        const float g  = zv / (1.f + __expf(-zv));           // silu(z)
        yv = (yv + uv * Dd) * g;
        y[token * dI + d] = (bf16)yv;
    }
}

// ---------------------------------------------------------------------------
// Host side
// ---------------------------------------------------------------------------
static inline void gemm(const bf16* A, const bf16* W, const float* bias,
                        bf16* oB, float* oF, int N, int K, int act,
                        hipStream_t st) {
    const dim3 blk(256);
    const int gy = NT_TOKENS / 128;
    if (N % 64 == 0) {
        dim3 grid(N / 64, gy);
        gemm_bf16_wmma<4><<<grid, blk, 0, st>>>(A, W, bias, oB, oF, N, K, act);
    } else if (N % 48 == 0) {           // N == 48
        dim3 grid(N / 48, gy);
        gemm_bf16_wmma<3><<<grid, blk, 0, st>>>(A, W, bias, oB, oF, N, K, act);
    } else {                            // N == 32
        dim3 grid(N / 32, gy);
        gemm_bf16_wmma<2><<<grid, blk, 0, st>>>(A, W, bias, oB, oF, N, K, act);
    }
}

extern "C" void kernel_launch(void* const* d_in, const int* in_sizes, int n_in,
                              void* d_out, int out_size, void* d_ws, size_t ws_size,
                              hipStream_t stream) {
    (void)in_sizes; (void)n_in; (void)out_size; (void)ws_size;
    // ---- input pointers (setup_inputs dict insertion order, recursive) ----
    const float* x      = (const float*)d_in[0];
    const float* cw4[4] = {(const float*)d_in[1], (const float*)d_in[3],
                           (const float*)d_in[5], (const float*)d_in[7]};
    const float* cb4[4] = {(const float*)d_in[2], (const float*)d_in[4],
                           (const float*)d_in[6], (const float*)d_in[8]};
    const float* fuse_w = (const float*)d_in[9];
    const float* fuse_b = (const float*)d_in[10];
    const int    base[2] = {11, 24};
    const int    dms[2]  = {256, 512};
    const float* flin1_w = (const float*)d_in[37];
    const float* flin1_b = (const float*)d_in[38];
    const float* flin2_w = (const float*)d_in[39];
    const float* flin2_b = (const float*)d_in[40];

    // ---- workspace carving ----
    char* wsp = (char*)d_ws;
    size_t off = 0;
    auto alloc = [&](size_t bytes) -> void* {
        void* p = wsp + off;
        off += (bytes + 255) & ~(size_t)255;
        return p;
    };
    auto cvt = [&](const float* src, size_t n) -> bf16* {
        bf16* dst = (bf16*)alloc(n * sizeof(bf16));
        f2bf_kernel<<<(int)((n + 255) / 256), 256, 0, stream>>>(src, dst, (int)n);
        return dst;
    };

    // bf16 weights
    bf16* fuse_wb = cvt(fuse_w, 256 * 256);
    bf16 *in_b[2], *xp_b[2], *op_b[2], *l1w[2], *l2w[2];
    for (int l = 0; l < 2; ++l) {
        const int dm = dms[l], dI = 2 * dm, dtr = dm / 16;
        in_b[l] = cvt((const float*)d_in[base[l] + 0], (size_t)2 * dI * dm);
        xp_b[l] = cvt((const float*)d_in[base[l] + 3], (size_t)(dtr + 32) * dI);
        op_b[l] = cvt((const float*)d_in[base[l] + 8], (size_t)dm * dI);
        l1w[l]  = cvt((const float*)d_in[base[l] + 9], (size_t)512 * dm);
        l2w[l]  = cvt((const float*)d_in[base[l] + 11], (size_t)512 * 512);
    }
    bf16* fl1 = cvt(flin1_w, (size_t)256 * 512);
    bf16* fl2 = cvt(flin2_w, (size_t)32 * 256);

    // activation buffers
    bf16*  act = (bf16*)alloc((size_t)NT_TOKENS * 512 * sizeof(bf16));
    bf16*  mid = (bf16*)alloc((size_t)NT_TOKENS * 512 * sizeof(bf16));
    bf16*  xzb = (bf16*)alloc((size_t)NT_TOKENS * 2048 * sizeof(bf16));
    bf16*  ub  = (bf16*)alloc((size_t)NT_TOKENS * 1024 * sizeof(bf16));
    bf16*  yb  = (bf16*)alloc((size_t)NT_TOKENS * 1024 * sizeof(bf16));
    float* xdb = (float*)alloc((size_t)NT_TOKENS * 64 * sizeof(float));

    // ---- front-end: dilated convs -> (NT,256) bf16, then fuse GEMM + relu ----
    dilated_conv_kernel<<<NT_TOKENS, 256, 0, stream>>>(
        x, cw4[0], cb4[0], cw4[1], cb4[1], cw4[2], cb4[2], cw4[3], cb4[3], mid);
    gemm(mid, fuse_wb, fuse_b, act, nullptr, 256, 256, /*relu*/1, stream);

    // ---- two mamba + FFN layers ----
    for (int l = 0; l < 2; ++l) {
        const int dm = dms[l], dI = 2 * dm, dtr = dm / 16, xw = dtr + 32;
        const float* conv_w = (const float*)d_in[base[l] + 1];
        const float* conv_b = (const float*)d_in[base[l] + 2];
        const float* dtw    = (const float*)d_in[base[l] + 4];
        const float* dtb    = (const float*)d_in[base[l] + 5];
        const float* Alog   = (const float*)d_in[base[l] + 6];
        const float* Dv     = (const float*)d_in[base[l] + 7];
        const float* l1b    = (const float*)d_in[base[l] + 10];
        const float* l2b    = (const float*)d_in[base[l] + 12];

        // xz = h @ in_proj.T   (NT, 2*dI)
        gemm(act, in_b[l], nullptr, xzb, nullptr, 2 * dI, dm, 0, stream);
        // u = silu(depthwise causal conv(xi))
        const int tot = NT_TOKENS * dI;
        if (dI == 512)
            dwconv_silu_kernel<512><<<(tot + 255) / 256, 256, 0, stream>>>(
                xzb, conv_w, conv_b, ub);
        else
            dwconv_silu_kernel<1024><<<(tot + 255) / 256, 256, 0, stream>>>(
                xzb, conv_w, conv_b, ub);
        // xdbl = u @ x_proj.T  (NT, dtr+32) f32
        gemm(ub, xp_b[l], nullptr, nullptr, xdb, xw, dI, 0, stream);
        // fused dt_proj + softplus + scan + skip + gate -> y (NT, dI) bf16
        if (dtr == 16)
            scan_kernel<16><<<BN_SEQS, dI, 0, stream>>>(
                xdb, xzb, ub, Alog, dtw, dtb, Dv, yb, dI);
        else
            scan_kernel<32><<<BN_SEQS, dI, 0, stream>>>(
                xdb, xzb, ub, Alog, dtw, dtb, Dv, yb, dI);
        // out_proj -> (NT, dm) ; FFN: relu(lin1) then lin2
        gemm(yb, op_b[l], nullptr, act, nullptr, dm, dI, 0, stream);
        gemm(act, l1w[l], l1b, mid, nullptr, 512, dm, 1, stream);
        gemm(mid, l2w[l], l2b, act, nullptr, 512, 512, 0, stream);
    }

    // ---- final head: relu(h @ lin1.T + b) @ lin2.T + b -> (NT, 32) f32 ----
    gemm(act, fl1, flin1_b, mid, nullptr, 256, 512, 1, stream);
    gemm(mid, fl2, flin2_b, nullptr, (float*)d_out, 32, 256, 0, stream);
}
